// LinearAttention_26749056319637
// MI455X (gfx1250) — matrix operand order repaired and checked by reading, not compiled, not run on hardware
//
#include <hip/hip_runtime.h>

#define B_ 4
#define S_ 2048
#define F_ 256
#define I_ 512
#define D_ 8
#define K_ 7
#define C_ 256

typedef __bf16 v16bf __attribute__((ext_vector_type(16)));
typedef __bf16 v8bf  __attribute__((ext_vector_type(8)));
typedef float  v8f   __attribute__((ext_vector_type(8)));
typedef float  v4f   __attribute__((ext_vector_type(4)));

// ---------------------------------------------------------------------------
// Async copy of one 16-byte chunk global -> LDS (CDNA5 async path, ASYNCcnt)
// ---------------------------------------------------------------------------
__device__ __forceinline__ void async_copy_b128(const __bf16* gsrc, __bf16* ldst) {
  unsigned lds_off = (unsigned)(unsigned long long)ldst;   // low 32 bits = LDS addr
  asm volatile("global_load_async_to_lds_b128 %0, %1, off"
               :: "v"(lds_off), "v"(gsrc)
               : "memory");
}
__device__ __forceinline__ void wait_async0() {
  asm volatile("s_wait_asynccnt 0x0" ::: "memory");
}
__device__ __forceinline__ void wait_async1() {           // all but newest done
  asm volatile("s_wait_asynccnt 0x1" ::: "memory");
}

// ---------------------------------------------------------------------------
// A fragment (16x32 bf16) from LDS panel row (contiguous k 0..31 per row):
//   j<8 : k = 8*(lane/16) + j ; j>=8 : k = 16 + 8*(lane/16) + (j-8)
// ---------------------------------------------------------------------------
__device__ __forceinline__ v16bf load_afrag_lds(const __bf16* rowp, int c0) {
  v16bf a;
  const v8bf a0 = *reinterpret_cast<const v8bf*>(rowp + c0);
  const v8bf a1 = *reinterpret_cast<const v8bf*>(rowp + 16 + c0);
#pragma unroll
  for (int j = 0; j < 8; ++j) { a[j] = a0[j]; a[8 + j] = a1[j]; }
  return a;
}

// ---------------------------------------------------------------------------
// B fragment (32x16 bf16) from TOKEN-MAJOR activations: lane = column (n = s),
//   j<8 : k = kk + 8*(lane/16) + j ; j>=8 : k = kk + 16 + 8*(lane/16) + (j-8)
// ---------------------------------------------------------------------------
__device__ __forceinline__ v16bf load_bfrag(const __bf16* __restrict__ rowp, int kk, int k0) {
  v16bf b;
  const v8bf b0 = *reinterpret_cast<const v8bf*>(rowp + kk + k0);
  const v8bf b1 = *reinterpret_cast<const v8bf*>(rowp + kk + k0 + 16);
#pragma unroll
  for (int j = 0; j < 8; ++j) { b[j] = b0[j]; b[8 + j] = b1[j]; }
  return b;
}

// ---------------------------------------------------------------------------
// Positional embedding, token-major: pos[s*F + f] = sin((s+1)*fe[f]) * 2^-0.5
// ---------------------------------------------------------------------------
__global__ void pos_kernel(float* __restrict__ pos) {
  int idx = blockIdx.x * blockDim.x + threadIdx.x;
  if (idx >= F_ * S_) return;
  int s = idx / F_;
  int f = idx % F_;
  const float PI = 3.14159265358979323846f;
  float fe0 = (float)f + 1.0f;
  float add = fmodf(fe0, 2.0f);
  float fe1 = (fe0 - add) * 0.5f;
  add *= PI;
  float fe2 = fe1 * (8.0f / (float)F_) - logf((float)C_ / (2.0f * PI));
  float fe  = expf(fe2) + add;
  pos[idx] = sinf(((float)s + 1.0f) * fe) * 0.70710678118654752f;
}

// ---------------------------------------------------------------------------
// fp32 -> bf16 bulk convert
// ---------------------------------------------------------------------------
__global__ void cvt_bf16_kernel(const float* __restrict__ src, __bf16* __restrict__ dst, long n) {
  long i = (long)blockIdx.x * blockDim.x + threadIdx.x;
  if (i < n) dst[i] = (__bf16)src[i];
}

// w1 (D,3,I,I,K) fp32 -> (D*3, K, O=I, I) bf16  (tap-major for conv GEMM)
__global__ void cvt_w1_kernel(const float* __restrict__ src, __bf16* __restrict__ dst) {
  long i = (long)blockIdx.x * blockDim.x + threadIdx.x;
  const long total = (long)D_ * 3 * K_ * I_ * I_;
  if (i >= total) return;
  long ii = i % I_;  long r = i / I_;
  long o  = r % I_;  r /= I_;
  long t  = r % K_;  long g = r / K_;   // g = d*3 + branch
  dst[i] = (__bf16)src[((g * I_ + o) * I_ + ii) * K_ + t];
}

// ---------------------------------------------------------------------------
// Embedding gather, token-major: v/x (B,S,F) fp32, obf = bf16(x + pos)
// ---------------------------------------------------------------------------
__global__ void embed_kernel(const float* __restrict__ emb, const int* __restrict__ inp,
                             const float* __restrict__ pos,
                             float* __restrict__ v, float* __restrict__ x,
                             __bf16* __restrict__ obf) {
  int idx = blockIdx.x * blockDim.x + threadIdx.x;   // (b,s) column
  if (idx >= B_ * S_) return;
  int s = idx % S_;
  long base = (long)idx * F_;
  const float* e = emb + (long)inp[idx] * (2 * F_);
  const float* ps = pos + (long)s * F_;
  for (int f = 0; f < F_; ++f) {
    float e0 = e[f];
    float e1 = e[F_ + f];
    v[base + f] = e0;
    x[base + f] = e1;
    obf[base + f] = (__bf16)(e1 + ps[f]);
  }
}

// ---------------------------------------------------------------------------
// WMMA GEMM, token-major activations:
//   Ct[b][n][m] = A[m][k] * Bt[b][n][k]
// block = 8 waves (2M x 4N), block tile 64M x 128N, wave tile 32x32, K step 32.
// A panel (64x32, 4KB) double-buffered in LDS via global_load_async_to_lds_b128
// (1 chunk per thread per k-step): issue panel i+1, s_wait_asynccnt 1 => panel
// i ready while i+1 overlaps the WMMAs.  B direct from global (token-major).
// Output channel folding: group g = m >> grpShift placed at g*grpStride.
// ---------------------------------------------------------------------------
__global__ __launch_bounds__(256)
void gemm_wmma_t(const __bf16* __restrict__ A, const __bf16* __restrict__ Bt,
                 float* __restrict__ Cf, __bf16* __restrict__ Cb,
                 int M, int Kd, long sBb, long sCb,
                 int grpShift, long grpStride, int ldc, int relu) {
  __shared__ __bf16 ldsA[2][64][40];            // 80B row stride, 16B aligned
  const int bz   = blockIdx.z;
  const int n0   = blockIdx.x * 128;
  const int m0   = blockIdx.y * 64;
  const int tid  = threadIdx.x;
  const int lane = tid & 31;
  const int wave = tid >> 5;
  const int wm   = wave >> 2;                   // 0..1
  const int wn   = wave & 3;                    // 0..3
  const int rowi = lane & 15;
  const int hi   = lane >> 4;
  const int c0   = hi * 8;                      // A k-offset pattern
  const int k0   = hi * 8;                      // B k-offset pattern
  const int BUFE = 64 * 40;                     // elements per LDS buffer

  // staging assignment: thread -> (row, 16B chunk) of the A panel
  const int smRow = tid >> 2;                   // 0..63
  const int smCh  = (tid & 3) * 8;              // k chunk base
  const __bf16* Astage = A + (long)(m0 + smRow) * Kd + smCh;
  __bf16* Adst0 = &ldsA[0][smRow][smCh];
  __bf16* Adst1 = &ldsA[1][smRow][smCh];

  const __bf16* ArowL0 = &ldsA[0][wm * 32 + rowi][0];
  const __bf16* ArowL1 = &ldsA[0][wm * 32 + 16 + rowi][0];

  const __bf16* Bbase = Bt + (long)bz * sBb;
  const __bf16* Br0 = Bbase + (long)(n0 + wn * 32 + rowi) * Kd;
  const __bf16* Br1 = Br0 + 16L * Kd;

  v8f acc00 = {}; v8f acc01 = {}; v8f acc10 = {}; v8f acc11 = {};

  async_copy_b128(Astage, Adst0);               // prologue: panel 0 -> buf 0
  for (int kk = 0; kk < Kd; kk += 32) {
    const int cur = (kk >> 5) & 1;
    if (kk + 32 < Kd) {
      async_copy_b128(Astage + kk + 32, cur ? Adst0 : Adst1);
      __builtin_prefetch(Br0 + kk + 32, 0, 1);
      __builtin_prefetch(Br1 + kk + 32, 0, 1);
      wait_async1();                            // panel kk done, kk+32 in flight
    } else {
      wait_async0();
    }
    __syncthreads();                            // all stagers' chunks visible
    const int bo = cur ? BUFE : 0;
    v16bf a0 = load_afrag_lds(ArowL0 + bo, c0);
    v16bf a1 = load_afrag_lds(ArowL1 + bo, c0);
    v16bf b0 = load_bfrag(Br0, kk, k0);
    v16bf b1 = load_bfrag(Br1, kk, k0);
    acc00 = __builtin_amdgcn_wmma_f32_16x16x32_bf16(false, a0, false, b0, (short)0, acc00, false, false);
    acc01 = __builtin_amdgcn_wmma_f32_16x16x32_bf16(false, a0, false, b1, (short)0, acc01, false, false);
    acc10 = __builtin_amdgcn_wmma_f32_16x16x32_bf16(false, a1, false, b0, (short)0, acc10, false, false);
    acc11 = __builtin_amdgcn_wmma_f32_16x16x32_bf16(false, a1, false, b1, (short)0, acc11, false, false);
    __syncthreads();                            // reads drained before overwrite
  }

  const int mmask = (1 << grpShift) - 1;
#pragma unroll
  for (int tm = 0; tm < 2; ++tm) {
#pragma unroll
    for (int tn = 0; tn < 2; ++tn) {
      v8f acc = tm == 0 ? (tn == 0 ? acc00 : acc01) : (tn == 0 ? acc10 : acc11);
      int mb = m0 + wm * 32 + tm * 16 + hi * 8;     // 8 consecutive out channels
      int n  = n0 + wn * 32 + tn * 16 + rowi;       // token index
      long off = (long)bz * sCb + (long)(mb >> grpShift) * grpStride
               + (long)n * ldc + (mb & mmask);
      float vals[8];
#pragma unroll
      for (int r = 0; r < 8; ++r) vals[r] = relu ? fmaxf(acc[r], 0.0f) : acc[r];
      if (Cb) {
        v8bf pk;
#pragma unroll
        for (int r = 0; r < 8; ++r) pk[r] = (__bf16)vals[r];
        *reinterpret_cast<v8bf*>(Cb + off) = pk;
      }
      if (Cf) {
        v4f lo, hiv;
#pragma unroll
        for (int r = 0; r < 4; ++r) { lo[r] = vals[r]; hiv[r] = vals[4 + r]; }
        *reinterpret_cast<v4f*>(Cf + off)     = lo;
        *reinterpret_cast<v4f*>(Cf + off + 4) = hiv;
      }
    }
  }
}

// ---------------------------------------------------------------------------
// Causal conv as 7 accumulated WMMA GEMMs (fixed shape M=Kd=I_): flattened
// (tap, k-step) pipeline of 112 iterations, double-buffered async A panels so
// the pipeline crosses tap boundaries.  Token shift = row offset t-6,
// out-of-range rows contribute zero fragments.  Always relu, bf16 out.
// ---------------------------------------------------------------------------
__global__ __launch_bounds__(256)
void conv_wmma_t(const __bf16* __restrict__ A7, const __bf16* __restrict__ Bt,
                 __bf16* __restrict__ Cb, long sBb, long sCb) {
  __shared__ __bf16 ldsA[2][64][40];
  const int bz   = blockIdx.z;
  const int n0   = blockIdx.x * 128;
  const int m0   = blockIdx.y * 64;
  const int tid  = threadIdx.x;
  const int lane = tid & 31;
  const int wave = tid >> 5;
  const int wm   = wave >> 2;
  const int wn   = wave & 3;
  const int rowi = lane & 15;
  const int hi   = lane >> 4;
  const int c0   = hi * 8;
  const int k0   = hi * 8;
  const int BUFE = 64 * 40;

  const int smRow = tid >> 2;
  const int smCh  = (tid & 3) * 8;
  const long rowOff = (long)(m0 + smRow) * I_ + smCh;
  __bf16* Adst0 = &ldsA[0][smRow][smCh];
  __bf16* Adst1 = &ldsA[1][smRow][smCh];
  const __bf16* ArowL0 = &ldsA[0][wm * 32 + rowi][0];
  const __bf16* ArowL1 = &ldsA[0][wm * 32 + 16 + rowi][0];

  const __bf16* Bbase = Bt + (long)bz * sBb;
  const int s0 = n0 + wn * 32 + rowi;           // token row, tn adds 16

  v8f acc00 = {}; v8f acc01 = {}; v8f acc10 = {}; v8f acc11 = {};

  const int KSTEPS = I_ >> 5;                   // 16 k-steps per tap
  const int TOTAL  = K_ * KSTEPS;               // 112 pipelined iterations
  // panel address for flattened iteration it = (tap, kstep)
  // tap = it >> 4, kk = (it & 15) << 5
  async_copy_b128(A7 + rowOff, Adst0);          // prologue: it = 0 -> buf 0
  for (int it = 0; it < TOTAL; ++it) {
    const int t   = it >> 4;
    const int kk  = (it & 15) << 5;
    const int cur = it & 1;
    if (it + 1 < TOTAL) {
      const int tn_ = (it + 1) >> 4;
      const int kn_ = ((it + 1) & 15) << 5;
      async_copy_b128(A7 + (long)tn_ * I_ * I_ + rowOff + kn_, cur ? Adst0 : Adst1);
      wait_async1();
    } else {
      wait_async0();
    }
    const int r0 = s0 + t - (K_ - 1);           // causal shift -6..0
    const int r1 = r0 + 16;
    const __bf16* Br0 = Bbase + (long)r0 * I_;
    const __bf16* Br1 = Bbase + (long)r1 * I_;
    __syncthreads();
    const int bo = cur ? BUFE : 0;
    v16bf a0 = load_afrag_lds(ArowL0 + bo, c0);
    v16bf a1 = load_afrag_lds(ArowL1 + bo, c0);
    v16bf b0, b1;
    if (r0 >= 0) b0 = load_bfrag(Br0, kk, k0); else b0 = (v16bf)(__bf16)0.0f;
    if (r1 >= 0) b1 = load_bfrag(Br1, kk, k0); else b1 = (v16bf)(__bf16)0.0f;
    acc00 = __builtin_amdgcn_wmma_f32_16x16x32_bf16(false, a0, false, b0, (short)0, acc00, false, false);
    acc01 = __builtin_amdgcn_wmma_f32_16x16x32_bf16(false, a0, false, b1, (short)0, acc01, false, false);
    acc10 = __builtin_amdgcn_wmma_f32_16x16x32_bf16(false, a1, false, b0, (short)0, acc10, false, false);
    acc11 = __builtin_amdgcn_wmma_f32_16x16x32_bf16(false, a1, false, b1, (short)0, acc11, false, false);
    __syncthreads();
  }

#pragma unroll
  for (int tm = 0; tm < 2; ++tm) {
#pragma unroll
    for (int tn = 0; tn < 2; ++tn) {
      v8f acc = tm == 0 ? (tn == 0 ? acc00 : acc01) : (tn == 0 ? acc10 : acc11);
      int mb = m0 + wm * 32 + tm * 16 + hi * 8;
      int n  = n0 + wn * 32 + tn * 16 + rowi;
      long off = (long)bz * sCb + (long)n * I_ + mb;
      v8bf pk;
#pragma unroll
      for (int r = 0; r < 8; ++r) pk[r] = (__bf16)fmaxf(acc[r], 0.0f);
      *reinterpret_cast<v8bf*>(Cb + off) = pk;
    }
  }
}

// ---------------------------------------------------------------------------
// Per-column channel pipeline (token-major, fully contiguous rows):
// dsh = (3, B, S, F): [0]=d (clobbered in place), [1]=sc, [2]=sh
// ---------------------------------------------------------------------------
__global__ void elem_kernel(float* __restrict__ dsh, float* __restrict__ v,
                            float* __restrict__ x, const float* __restrict__ pos,
                            __bf16* __restrict__ obf, __bf16* __restrict__ outcat,
                            int last) {
  int idx = blockIdx.x * blockDim.x + threadIdx.x;
  if (idx >= B_ * S_) return;
  int s = idx % S_;
  const long BSF = (long)B_ * S_ * F_;
  long base = (long)idx * F_;
  float*       dph = dsh + base;
  const float* scp = dsh + BSF + base;
  const float* shp = dsh + 2 * BSF + base;
  const float* ps  = pos + (long)s * F_;

  float invp = 1.0f / (float)(s + 1);
  float cum = 0.0f, sum = 0.0f, sumsq = 0.0f;
  for (int f = 0; f < F_; ++f) {
    cum += dph[f];
    float a = cum * invp * scp[f] + shp[f];
    dph[f] = a;
    sum += a;
    sumsq += a * a;
  }
  float mean  = sum / (float)F_;
  float nrm   = sqrtf(fmaxf(sumsq - (float)F_ * mean * mean, 0.0f));
  float denom = nrm * 0.0625f + 1e-5f;            // F^-0.5 = 1/16
  float rs    = 0.35355339059327373f / denom;     // init_scale = D^-0.5

  long cbase = (long)idx * (2 * F_);
  for (int f = 0; f < F_; ++f) {
    float att = (dph[f] - mean) * rs;
    float vn  = 0.99f * v[base + f] + 0.01f * att;
    float xn  = x[base + f] + vn;
    v[base + f] = vn;
    x[base + f] = xn;
    if (!last) {
      obf[base + f] = (__bf16)(xn + ps[f]);
    } else {
      outcat[cbase + f]      = (__bf16)vn;
      outcat[cbase + F_ + f] = (__bf16)xn;
    }
  }
}

// ---------------------------------------------------------------------------
// log-softmax NLL over token-major logits (B,S,C): one wave per column,
// class reads are lane-coalesced.
// ---------------------------------------------------------------------------
__global__ void nll_kernel(const float* __restrict__ logits, const float* __restrict__ out_b,
                           const int* __restrict__ tgt, float* __restrict__ nllbuf) {
  int gid  = blockIdx.x * blockDim.x + threadIdx.x;
  int col  = gid >> 5;
  int lane = gid & 31;
  if (col >= B_ * S_) return;
  const float* lg = logits + (long)col * C_;

  float lv[8];
  float lmax = -1e30f;
#pragma unroll
  for (int j = 0; j < 8; ++j) {
    int c = lane + j * 32;
    lv[j] = lg[c] + out_b[c];
    lmax = fmaxf(lmax, lv[j]);
  }
  for (int off = 16; off; off >>= 1) lmax = fmaxf(lmax, __shfl_xor(lmax, off, 32));
  float se = 0.0f;
#pragma unroll
  for (int j = 0; j < 8; ++j) se += expf(lv[j] - lmax);
  for (int off = 16; off; off >>= 1) se += __shfl_xor(se, off, 32);
  if (lane == 0) {
    int t = tgt[col];
    float lt = lg[t] + out_b[t];
    nllbuf[col] = (lmax + logf(se)) - lt;
  }
}

__global__ void reduce_kernel(const float* __restrict__ partial, float* __restrict__ out,
                              int n, float scale) {
  __shared__ float sm[256];
  float s = 0.0f;
  for (int i = threadIdx.x; i < n; i += 256) s += partial[i];
  sm[threadIdx.x] = s;
  __syncthreads();
  for (int st = 128; st; st >>= 1) {
    if (threadIdx.x < st) sm[threadIdx.x] += sm[threadIdx.x + st];
    __syncthreads();
  }
  if (threadIdx.x == 0) out[0] = sm[0] * scale;
}

// ---------------------------------------------------------------------------
extern "C" void kernel_launch(void* const* d_in, const int* in_sizes, int n_in,
                              void* d_out, int out_size, void* d_ws, size_t ws_size,
                              hipStream_t stream) {
  (void)in_sizes; (void)n_in; (void)out_size; (void)ws_size;
  const float* emb   = (const float*)d_in[0];
  const float* w0    = (const float*)d_in[1];
  const float* w1    = (const float*)d_in[2];
  const float* w2    = (const float*)d_in[3];
  const float* out_w = (const float*)d_in[4];
  const float* out_b = (const float*)d_in[5];
  const int*   inp   = (const int*)d_in[6];
  const int*   tgt   = (const int*)d_in[7];
  float*       out   = (float*)d_out;

  const long IF  = (long)I_ * F_;
  const long II  = (long)I_ * I_;
  const long FI  = (long)F_ * I_;
  const long FS  = (long)F_ * S_;        // token-major row block (S,F)
  const long IS  = (long)I_ * S_;
  const long BSF = (long)B_ * S_ * F_;

  // workspace carve-up (256-B aligned)
  char* p = (char*)d_ws;
  auto alloc = [&](size_t bytes) -> void* {
    void* r = (void*)p;
    p += (bytes + 255) & ~(size_t)255;
    return r;
  };
  __bf16* w0bf    = (__bf16*)alloc((size_t)D_ * 3 * IF * 2);
  __bf16* w1bf    = (__bf16*)alloc((size_t)D_ * 3 * K_ * II * 2);
  __bf16* w2bf    = (__bf16*)alloc((size_t)D_ * 3 * FI * 2);
  __bf16* owbf    = (__bf16*)alloc((size_t)C_ * 2 * F_ * 2);
  float*  pos     = (float*)alloc((size_t)S_ * F_ * 4);       // (S,F)
  float*  vbuf    = (float*)alloc((size_t)BSF * 4);           // (B,S,F)
  float*  xbuf    = (float*)alloc((size_t)BSF * 4);           // (B,S,F)
  __bf16* obf     = (__bf16*)alloc((size_t)BSF * 2);          // (B,S,F)
  __bf16* hbf     = (__bf16*)alloc((size_t)B_ * 3 * IS * 2);  // (B,3,S,I)
  __bf16* h2bf    = (__bf16*)alloc((size_t)B_ * 3 * IS * 2);  // (B,3,S,I)
  float*  dsh     = (float*)alloc((size_t)3 * BSF * 4);       // (3,B,S,F)
  __bf16* outcat  = (__bf16*)alloc((size_t)B_ * S_ * 2 * F_ * 2); // (B,S,2F)
  float*  logitsf = (float*)alloc((size_t)B_ * S_ * C_ * 4);  // (B,S,C)
  float*  nllbuf  = (float*)alloc((size_t)B_ * S_ * 4);

  // precompute / convert
  pos_kernel<<<(F_ * S_ + 255) / 256, 256, 0, stream>>>(pos);
  {
    long n0 = (long)D_ * 3 * IF;
    cvt_bf16_kernel<<<(unsigned)((n0 + 255) / 256), 256, 0, stream>>>(w0, w0bf, n0);
    long n2 = (long)D_ * 3 * FI;
    cvt_bf16_kernel<<<(unsigned)((n2 + 255) / 256), 256, 0, stream>>>(w2, w2bf, n2);
    long nw = (long)C_ * 2 * F_;
    cvt_bf16_kernel<<<(unsigned)((nw + 255) / 256), 256, 0, stream>>>(out_w, owbf, nw);
    long n1 = (long)D_ * 3 * K_ * II;
    cvt_w1_kernel<<<(unsigned)((n1 + 255) / 256), 256, 0, stream>>>(w1, w1bf);
  }
  embed_kernel<<<(B_ * S_ + 255) / 256, 256, 0, stream>>>(emb, inp, pos, vbuf, xbuf, obf);

  // layer loop
  for (int d = 0; d < D_; ++d) {
    // h = relu(W0 @ o): branches folded into M=3*I, group split at m>>9
    gemm_wmma_t<<<dim3(S_ / 128, (3 * I_) / 64, B_), 256, 0, stream>>>(
        w0bf + (long)d * 3 * IF, obf, (float*)nullptr, hbf,
        3 * I_, F_, FS, 3 * IS, /*grpShift=*/9, /*grpStride=*/IS, /*ldc=*/I_, /*relu=*/1);
    // h2 = relu(causal_conv(h)), per branch
    for (int br = 0; br < 3; ++br) {
      conv_wmma_t<<<dim3(S_ / 128, I_ / 64, B_), 256, 0, stream>>>(
          w1bf + ((long)(d * 3 + br)) * K_ * II, hbf + (long)br * IS,
          h2bf + (long)br * IS, 3 * IS, 3 * IS);
    }
    // d/sc/sh = W2 @ h2, per branch, fp32 out
    for (int br = 0; br < 3; ++br) {
      gemm_wmma_t<<<dim3(S_ / 128, F_ / 64, B_), 256, 0, stream>>>(
          w2bf + ((long)(d * 3 + br)) * FI, h2bf + (long)br * IS,
          dsh + (long)br * BSF, (__bf16*)nullptr,
          F_, I_, 3 * IS, FS, /*grpShift=*/30, /*grpStride=*/0, /*ldc=*/F_, /*relu=*/0);
    }
    elem_kernel<<<(B_ * S_ + 255) / 256, 256, 0, stream>>>(
        dsh, vbuf, xbuf, pos, obf, outcat, (d == D_ - 1) ? 1 : 0);
  }

  // logits = out_w @ concat(v,x)
  gemm_wmma_t<<<dim3(S_ / 128, C_ / 64, B_), 256, 0, stream>>>(
      owbf, outcat, logitsf, (__bf16*)nullptr,
      C_, 2 * F_, (long)S_ * 2 * F_, (long)S_ * C_,
      /*grpShift=*/30, /*grpStride=*/0, /*ldc=*/C_, /*relu=*/0);

  nll_kernel<<<(B_ * S_ * 32 + 255) / 256, 256, 0, stream>>>(logitsf, out_b, tgt, nllbuf);
  reduce_kernel<<<1, 256, 0, stream>>>(nllbuf, out, B_ * S_, 1.0f / (float)(B_ * S_));
}
